// Multihead_self_attention_11227044512277
// MI455X (gfx1250) — compile-verified, matrix operand order checked by reading, not run on hardware
//
#include <hip/hip_runtime.h>
#include <hip/hip_bf16.h>
#include <stdint.h>

// ---------------------------------------------------------------------------
// Multi-head self-attention, CDNA5 (gfx1250), bf16 WMMA pipeline.
//   Q/K/V/O projections: LDS-blocked 128x128 GEMM, TDM tensor_load_to_lds
//   (double-buffered, TENSORcnt-ordered) when available.
//   Attention: causal flash attention, online softmax in WMMA C-layout.
// ---------------------------------------------------------------------------

#define D_MODEL 1024
#define NH      16
#define BATCH   2
#define SEQ     2048
#define DKH     64                 // per-head dim
#define MROWS   (BATCH * SEQ)      // 4096

typedef __attribute__((ext_vector_type(16))) __bf16        v16bf;
typedef __attribute__((ext_vector_type(8)))  float         v8f;
typedef __attribute__((ext_vector_type(4)))  unsigned int  v4u;
typedef __attribute__((ext_vector_type(8)))  int           v8i;
typedef __attribute__((ext_vector_type(4)))  int           v4i;

#if defined(__gfx1250__) && __has_builtin(__builtin_amdgcn_tensor_load_to_lds) && \
    __has_builtin(__builtin_amdgcn_s_wait_tensorcnt)
#define USE_TDM 1
#else
#define USE_TDM 0
#endif

union FragBF {
    v16bf v;
    uint4 q[2];
};

// Per-lane A/B fragment load for 16x16x32 bf16 WMMA.
// Lane L covers K-offsets {base..base+7} and {base+16..base+23}, base=(L>=16)*8.
__device__ inline FragBF load_frag(const __hip_bfloat16* rowptr, int kchunk, int lane) {
    FragBF f;
    const int off = kchunk + ((lane >> 4) << 3);
    f.q[0] = *reinterpret_cast<const uint4*>(rowptr + off);
    f.q[1] = *reinterpret_cast<const uint4*>(rowptr + off + 16);
    return f;
}

__device__ inline v8f wmma_bf16(FragBF a, FragBF b, v8f c) {
    return __builtin_amdgcn_wmma_f32_16x16x32_bf16(
        /*neg_a=*/false, a.v, /*neg_b=*/false, b.v,
        /*c_mod=*/(short)0, c, /*reuse_a=*/false, /*reuse_b=*/false);
}

#if USE_TDM
// Issue one TDM 2D tile load: 128 rows x 32 bf16 columns, row stride D_MODEL.
// Descriptor per CDNA5 ISA (group0/group1; groups 2/3 zero).
__device__ inline void tdm_load_panel(unsigned long long ga, unsigned lds_off) {
    v4u g0 = { 1u,                                    // count=1, user descriptor
               lds_off,                               // lds_addr [63:32]
               (unsigned)(ga & 0xFFFFFFFFull),        // global_addr[31:0]
               (unsigned)((ga >> 32) & 0x01FFFFFFull) // global_addr[56:32]
                   | 0x80000000u };                   // type=2 ("image")
    v8i g1 = { (int)0x00010000,                       // data_size=2B, mask=0
               (int)((D_MODEL & 0xFFFF) << 16),       // tensor_dim0[15:0]
               (int)(((unsigned)D_MODEL >> 16) |
                     ((128u & 0xFFFFu) << 16)),       // dim0 hi | tensor_dim1 lo
               (int)((128u >> 16) | (32u << 16)),     // dim1 hi | tile_dim0=32
               (int)128,                              // tile_dim1=128, tile_dim2=0
               (int)D_MODEL,                          // tensor_dim0_stride lo
               0, 0 };                                // stride hi, dim1_stride
    v4i z4 = { 0, 0, 0, 0 };
#if __clang_major__ >= 23
    v8i z8 = { 0, 0, 0, 0, 0, 0, 0, 0 };
    __builtin_amdgcn_tensor_load_to_lds(g0, g1, z4, z4, z8, 0);
#else
    __builtin_amdgcn_tensor_load_to_lds(g0, g1, z4, z4, 0);
#endif
}
#endif

// ---------------------------------------------------------------------------
// Kernel 0: f32 -> bf16 conversion (grid-stride)
// ---------------------------------------------------------------------------
__global__ void cvt_f32_bf16(const float* __restrict__ in,
                             __hip_bfloat16* __restrict__ out, int n) {
    int i = blockIdx.x * blockDim.x + threadIdx.x;
    const int stride = gridDim.x * blockDim.x;
    for (; i < n; i += stride) out[i] = __float2bfloat16(in[i]);
}

// ---------------------------------------------------------------------------
// One K-step of the block GEMM: each wave reads its fragments from the LDS
// panels and performs a 2x4 grid of 16x16x32 WMMAs (32x64 register tile).
// ---------------------------------------------------------------------------
__device__ inline void mma_step(const __hip_bfloat16* __restrict__ Asb,
                                const __hip_bfloat16* __restrict__ Bsb,
                                v8f acc[2][4], int wm, int wn, int l15, int koff) {
    FragBF a[2], b[4];
    #pragma unroll
    for (int t = 0; t < 2; ++t) {
        const __hip_bfloat16* p = Asb + (wm * 32 + t * 16 + l15) * 32 + koff;
        a[t].q[0] = *reinterpret_cast<const uint4*>(p);
        a[t].q[1] = *reinterpret_cast<const uint4*>(p + 16);
    }
    #pragma unroll
    for (int t = 0; t < 4; ++t) {
        const __hip_bfloat16* p = Bsb + (wn * 64 + t * 16 + l15) * 32 + koff;
        b[t].q[0] = *reinterpret_cast<const uint4*>(p);
        b[t].q[1] = *reinterpret_cast<const uint4*>(p + 16);
    }
    #pragma unroll
    for (int i = 0; i < 2; ++i)
        #pragma unroll
        for (int j = 0; j < 4; ++j)
            acc[i][j] = wmma_bf16(a[i], b[j], acc[i][j]);
}

// ---------------------------------------------------------------------------
// Kernel 1: block-tiled GEMM  Y[m,n] = sum_k A[m,k] * W[n,k]
// A: [4096,1024] bf16 row-major.  W: [1024,1024] bf16 row-major.
// Block = 256 threads = 8 waves, owns 128x128 of Y. K-steps of 32 staged in
// LDS (TDM double-buffered when available). Wave (wm,wn) owns rows
// wm*32..+31, cols wn*64..+63.
// mode 0: Q  -> bf16 head-major [B,H,S,64]
// mode 1: K  -> bf16 head-major + f32 head-major (d_out K slot)
// mode 2: V  -> bf16 transposed [B,H,64,S] + f32 head-major (d_out V slot)
// mode 3: O  -> f32 plain [B,S,D] (d_out out slot)
// ---------------------------------------------------------------------------
__global__ __launch_bounds__(256)
void gemm_block128(const __hip_bfloat16* __restrict__ A,
                   const __hip_bfloat16* __restrict__ W,
                   __hip_bfloat16* __restrict__ outB,
                   float* __restrict__ outF,
                   int mode) {
    __shared__ __align__(16) __hip_bfloat16 As[2][128 * 32];
    __shared__ __align__(16) __hip_bfloat16 Bs[2][128 * 32];

    const int lane  = threadIdx.x & 31;
    const int wave  = threadIdx.x >> 5;
    const int bn    = blockIdx.x & 7;          // N/128 = 8
    const int bm    = blockIdx.x >> 3;         // M/128 = 32
    const int wm    = wave >> 1;               // 0..3
    const int wn    = wave & 1;                // 0..1
    const int l15   = lane & 15;
    const int koff  = (lane >> 4) << 3;        // 0 or 8
    const int half8 = (lane >> 4) << 3;

    const __hip_bfloat16* Ag = A + (size_t)bm * 128 * D_MODEL;
    const __hip_bfloat16* Wg = W + (size_t)bn * 128 * D_MODEL;

    v8f acc[2][4] = {};

#if USE_TDM
    const unsigned ldsA[2] = { (unsigned)(size_t)&As[0][0], (unsigned)(size_t)&As[1][0] };
    const unsigned ldsB[2] = { (unsigned)(size_t)&Bs[0][0], (unsigned)(size_t)&Bs[1][0] };
    if (wave == 0) {
        tdm_load_panel((unsigned long long)(size_t)Ag, ldsA[0]);
        tdm_load_panel((unsigned long long)(size_t)Wg, ldsB[0]);
    }
    int buf = 0;
    for (int kk = 0; kk < D_MODEL; kk += 32) {
        if (wave == 0) {
            const int kn = kk + 32;
            if (kn < D_MODEL) {
                tdm_load_panel((unsigned long long)(size_t)(Ag + kn), ldsA[buf ^ 1]);
                tdm_load_panel((unsigned long long)(size_t)(Wg + kn), ldsB[buf ^ 1]);
                __builtin_amdgcn_s_wait_tensorcnt(2);  // older pair retired
            } else {
                __builtin_amdgcn_s_wait_tensorcnt(0);
            }
        }
        __syncthreads();
        mma_step(&As[buf][0], &Bs[buf][0], acc, wm, wn, l15, koff);
        __syncthreads();
        buf ^= 1;
    }
#else
    const int trow = threadIdx.x >> 1;          // 0..127
    const int tcol = (threadIdx.x & 1) * 16;    // 0 or 16
    for (int kk = 0; kk < D_MODEL; kk += 32) {
        __syncthreads();
        *reinterpret_cast<uint4*>(&As[0][trow * 32 + tcol]) =
            *reinterpret_cast<const uint4*>(Ag + (size_t)trow * D_MODEL + kk + tcol);
        *reinterpret_cast<uint4*>(&Bs[0][trow * 32 + tcol]) =
            *reinterpret_cast<const uint4*>(Wg + (size_t)trow * D_MODEL + kk + tcol);
        __syncthreads();
        mma_step(&As[0][0], &Bs[0][0], acc, wm, wn, l15, koff);
    }
#endif

    // ---- epilogue: scatter per mode ----
    #pragma unroll
    for (int tm = 0; tm < 2; ++tm) {
        #pragma unroll
        for (int tn = 0; tn < 4; ++tn) {
            #pragma unroll
            for (int r = 0; r < 8; ++r) {
                const int m = bm * 128 + wm * 32 + tm * 16 + r + half8;
                const int n = bn * 128 + wn * 64 + tn * 16 + l15;
                const float val = acc[tm][tn][r];
                if (mode == 3) {
                    outF[(size_t)m * D_MODEL + n] = val;
                } else {
                    const int b = m >> 11;             // / SEQ
                    const int s = m & (SEQ - 1);
                    const int h = n >> 6;              // / DKH
                    const int d = n & (DKH - 1);
                    const size_t hm = (((size_t)(b * NH + h) * SEQ) + s) * DKH + d;
                    if (mode == 0) {
                        outB[hm] = __float2bfloat16(val);
                    } else if (mode == 1) {
                        outB[hm] = __float2bfloat16(val);
                        outF[hm] = val;
                    } else {                           // mode 2: V (also transposed)
                        const size_t tv = (((size_t)(b * NH + h) * DKH) + d) * SEQ + s;
                        outB[tv] = __float2bfloat16(val);
                        outF[hm] = val;
                    }
                }
            }
        }
    }
}

// ---------------------------------------------------------------------------
// Kernel 2: causal flash attention, 1 wave per 16-query tile.
// Qb, Kb: [B,H,S,64] bf16.  Vt: [B,H,64,S] bf16.  Ob: merged [B,S,D] bf16.
// Softmax kept in WMMA C-layout (lane owns rows r+8*(lane/16)); row
// reductions via shfl_xor within 16-lane halves. P is transposed C-layout ->
// A-layout through LDS, ordered with an explicit s_wait_dscnt.
// ---------------------------------------------------------------------------
__global__ __launch_bounds__(128)
void attn_flash(const __hip_bfloat16* __restrict__ Qb,
                const __hip_bfloat16* __restrict__ Kb,
                const __hip_bfloat16* __restrict__ Vt,
                __hip_bfloat16* __restrict__ Ob) {
    __shared__ __align__(16) __hip_bfloat16 sP[4][16 * 32];

    const int lane  = threadIdx.x & 31;
    const int wave  = threadIdx.x >> 5;
    const int qt    = blockIdx.x * 4 + wave;  // 0..4095
    const int qi    = qt & 127;
    const int h     = (qt >> 7) & (NH - 1);
    const int b     = qt >> 11;
    const int qbase = qi * 16;
    const int half8 = (lane >> 4) << 3;       // 0 or 8
    const int l15   = lane & 15;

    const __hip_bfloat16* qhead = Qb + (size_t)(b * NH + h) * SEQ * DKH;
    const __hip_bfloat16* khead = Kb + (size_t)(b * NH + h) * SEQ * DKH;
    const __hip_bfloat16* vhead = Vt + (size_t)(b * NH + h) * DKH * SEQ;

    const __hip_bfloat16* qrow = qhead + (size_t)(qbase + l15) * DKH;
    const FragBF qf0 = load_frag(qrow, 0,  lane);
    const FragBF qf1 = load_frag(qrow, 32, lane);

    v8f o0 = {}, o1 = {}, o2 = {}, o3 = {};
    float mrow[8], lrow[8];
    #pragma unroll
    for (int r = 0; r < 8; ++r) { mrow[r] = -1e30f; lrow[r] = 0.0f; }
    const float scale = 0.125f;               // 1/sqrt(64)

    for (int kb = 0; kb <= qbase + 15; kb += 32) {
        // ---- scores: 16 queries x 32 keys (two 16-key N-tiles) ----
        v8f sc0 = {}, sc1 = {};
        {
            const __hip_bfloat16* krow0 = khead + (size_t)(kb + l15) * DKH;
            FragBF k00 = load_frag(krow0, 0, lane);
            FragBF k01 = load_frag(krow0, 32, lane);
            sc0 = wmma_bf16(qf0, k00, sc0);
            sc0 = wmma_bf16(qf1, k01, sc0);
            const __hip_bfloat16* krow1 = krow0 + 16 * DKH;
            FragBF k10 = load_frag(krow1, 0, lane);
            FragBF k11 = load_frag(krow1, 32, lane);
            sc1 = wmma_bf16(qf0, k10, sc1);
            sc1 = wmma_bf16(qf1, k11, sc1);
        }

        // ---- causal mask + online softmax (C-layout) ----
        #pragma unroll
        for (int r = 0; r < 8; ++r) {
            const int qg   = qbase + r + half8;
            const int key0 = kb + l15;
            const int key1 = key0 + 16;
            float s0 = sc0[r] * scale;
            float s1 = sc1[r] * scale;
            if (key0 > qg) s0 = -1e30f;
            if (key1 > qg) s1 = -1e30f;

            float rmax = fmaxf(s0, s1);
            #pragma unroll
            for (int msk = 8; msk; msk >>= 1)
                rmax = fmaxf(rmax, __shfl_xor(rmax, msk, 32));

            const float mnew  = fmaxf(mrow[r], rmax);
            const float alpha = __expf(mrow[r] - mnew);
            const float p0 = __expf(s0 - mnew);
            const float p1 = __expf(s1 - mnew);
            float rs = p0 + p1;
            #pragma unroll
            for (int msk = 8; msk; msk >>= 1)
                rs += __shfl_xor(rs, msk, 32);

            lrow[r] = lrow[r] * alpha + rs;
            mrow[r] = mnew;
            o0[r] *= alpha; o1[r] *= alpha; o2[r] *= alpha; o3[r] *= alpha;

            const int lr = r + half8;
            sP[wave][lr * 32 + l15]      = __float2bfloat16(p0);
            sP[wave][lr * 32 + 16 + l15] = __float2bfloat16(p1);
        }

        // LDS is in-order within a wave; enforce store->load ordering.
        asm volatile("s_wait_dscnt 0" ::: "memory");

        // ---- reload P as A-fragment: row = lane%16, 32 keys ----
        FragBF pf;
        {
            const __hip_bfloat16* prow = &sP[wave][l15 * 32];
            const int off = (lane >> 4) << 3;
            pf.q[0] = *reinterpret_cast<const uint4*>(prow + off);
            pf.q[1] = *reinterpret_cast<const uint4*>(prow + off + 16);
        }

        // ---- O += P (16x32) x V (32x64): four 16-wide N-tiles ----
        {
            const __hip_bfloat16* v0 = vhead + (size_t)(0 * 16 + l15) * SEQ;
            const __hip_bfloat16* v1 = vhead + (size_t)(1 * 16 + l15) * SEQ;
            const __hip_bfloat16* v2 = vhead + (size_t)(2 * 16 + l15) * SEQ;
            const __hip_bfloat16* v3 = vhead + (size_t)(3 * 16 + l15) * SEQ;
            o0 = wmma_bf16(pf, load_frag(v0, kb, lane), o0);
            o1 = wmma_bf16(pf, load_frag(v1, kb, lane), o1);
            o2 = wmma_bf16(pf, load_frag(v2, kb, lane), o2);
            o3 = wmma_bf16(pf, load_frag(v3, kb, lane), o3);
        }
    }

    // ---- normalize + store merged [B,S,D] bf16 ----
    #pragma unroll
    for (int r = 0; r < 8; ++r) {
        const float inv = 1.0f / lrow[r];
        const int q = qbase + r + half8;
        const size_t base = ((size_t)b * SEQ + q) * D_MODEL + h * DKH;
        Ob[base + 0  + l15] = __float2bfloat16(o0[r] * inv);
        Ob[base + 16 + l15] = __float2bfloat16(o1[r] * inv);
        Ob[base + 32 + l15] = __float2bfloat16(o2[r] * inv);
        Ob[base + 48 + l15] = __float2bfloat16(o3[r] * inv);
    }
}

// ---------------------------------------------------------------------------
extern "C" void kernel_launch(void* const* d_in, const int* in_sizes, int n_in,
                              void* d_out, int out_size, void* d_ws, size_t ws_size,
                              hipStream_t stream) {
    const float* X  = (const float*)d_in[0];
    const float* Wq = (const float*)d_in[1];
    const float* Wk = (const float*)d_in[2];
    const float* Wv = (const float*)d_in[3];
    const float* Wo = (const float*)d_in[4];
    float* out = (float*)d_out;

    const size_t NX = (size_t)MROWS * D_MODEL;      // 4,194,304 (== B*H*S*dk)
    const size_t NW = (size_t)D_MODEL * D_MODEL;    // 1,048,576

    // Workspace carve-out (bf16): 48 MiB total.
    __hip_bfloat16* Xb  = (__hip_bfloat16*)d_ws;
    __hip_bfloat16* Wqb = Xb  + NX;
    __hip_bfloat16* Wkb = Wqb + NW;
    __hip_bfloat16* Wvb = Wkb + NW;
    __hip_bfloat16* Wob = Wvb + NW;
    __hip_bfloat16* Qb  = Wob + NW;   // [B,H,S,64]
    __hip_bfloat16* Kb  = Qb  + NX;   // [B,H,S,64]
    __hip_bfloat16* Vt  = Kb  + NX;   // [B,H,64,S]
    __hip_bfloat16* Ab  = Vt  + NX;   // merged attn [B,S,D]

    float* outK = out + NX;           // d_out layout: out | K | V
    float* outV = outK + NX;

    cvt_f32_bf16<<<2048, 256, 0, stream>>>(X,  Xb,  (int)NX);
    cvt_f32_bf16<<<512,  256, 0, stream>>>(Wq, Wqb, (int)NW);
    cvt_f32_bf16<<<512,  256, 0, stream>>>(Wk, Wkb, (int)NW);
    cvt_f32_bf16<<<512,  256, 0, stream>>>(Wv, Wvb, (int)NW);
    cvt_f32_bf16<<<512,  256, 0, stream>>>(Wo, Wob, (int)NW);

    const int gemmBlocks = (MROWS / 128) * (D_MODEL / 128);   // 256
    gemm_block128<<<gemmBlocks, 256, 0, stream>>>(Xb, Wqb, Qb, nullptr, 0);
    gemm_block128<<<gemmBlocks, 256, 0, stream>>>(Xb, Wkb, Kb, outK,    1);
    gemm_block128<<<gemmBlocks, 256, 0, stream>>>(Xb, Wvb, Vt, outV,    2);

    attn_flash<<<BATCH * NH * (SEQ / 16) / 4, 128, 0, stream>>>(Qb, Kb, Vt, Ab);

    gemm_block128<<<gemmBlocks, 256, 0, stream>>>(Ab, Wob, nullptr, out, 3);
}
